// LatentGuidedEdgeDecoder_22565758174063
// MI455X (gfx1250) — matrix operand order repaired and checked by reading, not compiled, not run on hardware
//
#include <hip/hip_runtime.h>
#include <math.h>

// ---------------------------------------------------------------------------
// LatentGuidedEdgeDecoder forward, fused, for gfx1250 (MI455X, wave32, WMMA).
//   H=256, LAT=8, NC=8, B=65536.
// v3: weights pre-packed f32->bf16 into native WMMA B-fragment layout (static
// __device__ arena, ~2.1MB, L2-resident). Main-kernel GEMMs process TWO output
// n-tiles per A-fragment load (dual accumulators), so the steady state is:
//   1x ds_load_b128-pair (A, shared) + 2x global_load_b128-pair (B) + 2x wmma.
// Each wave owns a 16-row batch tile; 4 waves/block so weight streams dedup
// through WGP$/L2.
// ---------------------------------------------------------------------------

typedef __attribute__((ext_vector_type(16))) __bf16 v16bf;
typedef __attribute__((ext_vector_type(8)))  float  v8f;

struct alignas(16) U4 { unsigned x, y, z, w; };
union AFrag { U4 q[2]; v16bf v; };

#define TILE_ELEMS 512          // bf16 elements per packed 16x32 tile (1 KB)
#define SY_STRIDE  260          // f32 stride of LN staging (bank-conflict-free)
#define WAVE_SLICE 66560        // bytes of LDS per wave (24K + 24K + 16.25K pad)
#define OFF_SA1    0
#define OFF_SFUS   24576
#define OFF_SY     49152

// Packed bf16 weight arena (all GEMM B operands, padded, fragment layout).
#define WOFF_We   0         // 512x256  -> 131072
#define WOFF_Wc   131072    // 32x256 (K padded from 8)  -> 8192
#define WOFF_Wv   139264    // 256x256  -> 65536
#define WOFF_Wo   204800    // 256x256  -> 65536
#define WOFF_Wih  270336    // 512x768  -> 393216
#define WOFF_Whh  663552    // 256x768  -> 196608
#define WOFF_Wf   860160    // 768x256  -> 196608
#define WOFF_W1   1056768   // 256x128  -> 32768
#define WOFF_W2   1089536   // 128x16 (N padded from 8)  -> 2048
#define WPACK_TOTAL 1091584

__device__ __attribute__((aligned(16))) __bf16 g_wpack[WPACK_TOTAL];

// --------------------------- fragment layouts ------------------------------
// A operand (16x32, 16-bit; ISA 7.12.2): lane = m + 16*half;
//   v0..v3: K = half*8 + {0..7}; v4..v7: K = 16 + half*8 + {0..7}.
__device__ __forceinline__ int a_off(int m, int c) {
  int tile = c >> 5, cc = c & 31;
  int grp = cc >> 3;
  int halfA = grp & 1;
  int v = ((grp >> 1) << 2) + ((cc & 7) >> 1);
  return tile * TILE_ELEMS + (m + (halfA << 4)) * 16 + v * 2 + (cc & 1);
}

// B operand (32x16, 16-bit; same K order as A): lane = n + 16*half;
//   v0..v3: K = half*8 + {0..7}; v4..v7: K = 16 + half*8 + {0..7}.
// Tiles ordered n-major so a GEMM's kt-loop reads contiguous 1KB tiles.
__device__ __forceinline__ int b_off(int k, int n, int Ktiles) {
  int halfB = (k >> 3) & 1;
  int v = (((k >> 4) & 1) << 2) + ((k & 7) >> 1);
  int tile = (n >> 4) * Ktiles + (k >> 5);
  return tile * TILE_ELEMS + ((n & 15) + (halfB << 4)) * 16 + v * 2 + (k & 1);
}

// Prep: scatter f32 weight [K,N] (zero-padded to [Kp,Np]) into fragment layout.
__global__ void pack_weight_kernel(const float* __restrict__ src, int K, int N,
                                   int Kp, int Np, int dstOff) {
  int idx = blockIdx.x * 256 + threadIdx.x;
  if (idx >= Kp * Np) return;
  int k = idx / Np, n = idx % Np;
  float v = (k < K && n < N) ? src[(long)k * N + n] : 0.f;
  g_wpack[dstOff + b_off(k, n, Kp >> 5)] = (__bf16)v;
}

// ------------------------------- helpers -----------------------------------
__device__ __forceinline__ v8f splat8(float x) {
  v8f v;
#pragma unroll
  for (int i = 0; i < 8; ++i) v[i] = x;
  return v;
}

__device__ __forceinline__ AFrag load_frag128(const __bf16* base, int lane) {
  AFrag a;
  const U4* p = (const U4*)base;
  a.q[0] = p[lane * 2];
  a.q[1] = p[lane * 2 + 1];
  return a;
}

// Dual-tile GEMM: one shared A fragment feeds two B streams / accumulators.
template<int KT>
__device__ __forceinline__ void gemm_acc2(v8f& acc0, v8f& acc1,
                                          const __bf16* aT,
                                          const __bf16* __restrict__ w0,
                                          const __bf16* __restrict__ w1,
                                          int lane) {
  for (int kt = 0; kt < KT; ++kt) {
    AFrag a  = load_frag128(aT + kt * TILE_ELEMS, lane);
    AFrag b0 = load_frag128(w0 + kt * TILE_ELEMS, lane);
    AFrag b1 = load_frag128(w1 + kt * TILE_ELEMS, lane);
    acc0 = __builtin_amdgcn_wmma_f32_16x16x32_bf16(false, a.v, false, b0.v,
                                                   (short)0, acc0, false, false);
    acc1 = __builtin_amdgcn_wmma_f32_16x16x32_bf16(false, a.v, false, b1.v,
                                                   (short)0, acc1, false, false);
  }
}

// Single-tile GEMM (used where only one n-tile exists).
template<int KT>
__device__ __forceinline__ v8f gemm_acc(v8f acc, const __bf16* aT,
                                        const __bf16* __restrict__ wT, int lane) {
  for (int kt = 0; kt < KT; ++kt) {
    AFrag a = load_frag128(aT + kt * TILE_ELEMS, lane);
    AFrag b = load_frag128(wT + kt * TILE_ELEMS, lane);
    acc = __builtin_amdgcn_wmma_f32_16x16x32_bf16(false, a.v, false, b.v,
                                                  (short)0, acc, false, false);
  }
  return acc;
}

__device__ __forceinline__ void acc_to_sY(float* sY, v8f acc, int n0, int lane) {
  int nl = lane & 15, half = lane >> 4;
#pragma unroll
  for (int r = 0; r < 8; ++r) sY[(r + half * 8) * SY_STRIDE + n0 + nl] = acc[r];
}

__device__ __forceinline__ void acc_to_packed(__bf16* dst, v8f acc, int n0,
                                              bool relu, int lane) {
  int nl = lane & 15, half = lane >> 4;
#pragma unroll
  for (int r = 0; r < 8; ++r) {
    float y = acc[r];
    if (relu) y = fmaxf(y, 0.f);
    dst[a_off(r + half * 8, n0 + nl)] = (__bf16)y;
  }
}

// Row-wise LayerNorm over 256 cols from sY; pack bf16 into one/two A buffers.
__device__ __forceinline__ void wave_ln_pack(const float* sY,
                                             const float* __restrict__ g,
                                             const float* __restrict__ bb,
                                             __bf16* dst1, __bf16* dst2,
                                             bool relu, int lane) {
  int row = lane >> 1, sub = lane & 1;
  float s = 0.f, ss = 0.f;
  for (int i = 0; i < 128; ++i) {
    float x = sY[row * SY_STRIDE + 2 * i + sub];
    s += x; ss += x * x;
  }
  s  += __shfl_xor(s, 1);
  ss += __shfl_xor(ss, 1);
  float mean = s * (1.f / 256.f);
  float var  = ss * (1.f / 256.f) - mean * mean;
  float rstd = rsqrtf(var + 1e-5f);
  for (int i = 0; i < 128; ++i) {
    int c = 2 * i + sub;
    float x = sY[row * SY_STRIDE + c];
    float y = (x - mean) * rstd * g[c] + bb[c];
    if (relu) y = fmaxf(y, 0.f);
    __bf16 h = (__bf16)y;
    dst1[a_off(row, c)] = h;
    if (dst2) dst2[a_off(row, c)] = h;
  }
}

__device__ __forceinline__ void pack_rows256(__bf16* dst,
                                             const float* __restrict__ src,
                                             int lane) {
  for (int idx = lane; idx < 4096; idx += 32)
    dst[a_off(idx >> 8, idx & 255)] = (__bf16)src[idx];
}

__device__ __forceinline__ float sigm(float x) { return 1.f / (1.f + expf(-x)); }

// ------------------------------ main kernel --------------------------------
__global__ __launch_bounds__(128) void edge_decoder_kernel(
    const float* __restrict__ node_i, const float* __restrict__ node_j,
    const float* __restrict__ latent, const float* __restrict__ ehs,
    const int*   __restrict__ tok,
    const float* __restrict__ be,  const float* __restrict__ ge,
    const float* __restrict__ bee, const float* __restrict__ bc,
    const float* __restrict__ bvv, const float* __restrict__ bo,
    const float* __restrict__ gn,  const float* __restrict__ bn,
    const float* __restrict__ emb,
    const float* __restrict__ bih, const float* __restrict__ bhh,
    const float* __restrict__ bfp, const float* __restrict__ gf,
    const float* __restrict__ bff, const float* __restrict__ b1,
    const float* __restrict__ b2,
    float* __restrict__ out_logits, float* __restrict__ out_newh, int Bn) {
  const int lane = threadIdx.x & 31;
  const int wave = threadIdx.x >> 5;
  const int R = (blockIdx.x * 4 + wave) * 16;
  if (R + 16 > Bn) return;

  extern __shared__ char smem[];
  char* wb = smem + (size_t)wave * WAVE_SLICE;
  __bf16* sA1  = (__bf16*)(wb + OFF_SA1);   // 24 tiles: GEMM input staging
  __bf16* sFus = (__bf16*)(wb + OFF_SFUS);  // 24 tiles: [edge|attended|new_h]
  float*  sY   = (float*)(wb + OFF_SY);     // 16 x 260 f32 LN staging

  const int nl = lane & 15, half = lane >> 4;
  const __bf16* __restrict__ gW = g_wpack;

  // ---- Stage 1: edge = relu(LN(cat(node_i,node_j) @ We + be)) ----
  pack_rows256(sA1,                  node_i + (long)R * 256, lane);
  pack_rows256(sA1 + 8 * TILE_ELEMS, node_j + (long)R * 256, lane);
  for (int p = 0; p < 8; ++p) {
    int nt = 2 * p;
    v8f a0 = splat8(be[nt * 16 + nl]);
    v8f a1 = splat8(be[(nt + 1) * 16 + nl]);
    gemm_acc2<16>(a0, a1, sA1,
                  gW + WOFF_We + (nt)     * 16 * TILE_ELEMS,
                  gW + WOFF_We + (nt + 1) * 16 * TILE_ELEMS, lane);
    acc_to_sY(sY, a0, nt * 16, lane);
    acc_to_sY(sY, a1, (nt + 1) * 16, lane);
  }
  wave_ln_pack(sY, ge, bee, sFus, nullptr, true, lane);   // edge -> sFus[0..7]

  // ---- Stage 2: context = latent @ Wc + bc (K=8, zero-padded to 32) ----
  {
    AFrag f; f.q[0] = {0, 0, 0, 0}; f.q[1] = {0, 0, 0, 0};
    if (lane < 16) {
      const float* lp = latent + (long)(R + lane) * 8;
#pragma unroll
      for (int j = 0; j < 8; ++j) f.v[j] = (__bf16)lp[j];  // cc 0..7, lanes 0..15
    }
    U4* p = (U4*)(sA1 + 16 * TILE_ELEMS);
    p[lane * 2] = f.q[0]; p[lane * 2 + 1] = f.q[1];
  }
  for (int p = 0; p < 8; ++p) {
    int nt = 2 * p;
    v8f a0 = splat8(bc[nt * 16 + nl]);
    v8f a1 = splat8(bc[(nt + 1) * 16 + nl]);
    gemm_acc2<1>(a0, a1, sA1 + 16 * TILE_ELEMS,
                 gW + WOFF_Wc + (nt)     * TILE_ELEMS,
                 gW + WOFF_Wc + (nt + 1) * TILE_ELEMS, lane);
    acc_to_packed(sA1, a0, nt * 16, false, lane);          // context -> sA1[0..7]
    acc_to_packed(sA1, a1, (nt + 1) * 16, false, lane);
  }

  // ---- Stage 3: v = context @ Wv + bv ----
  for (int p = 0; p < 8; ++p) {
    int nt = 2 * p;
    v8f a0 = splat8(bvv[nt * 16 + nl]);
    v8f a1 = splat8(bvv[(nt + 1) * 16 + nl]);
    gemm_acc2<8>(a0, a1, sA1,
                 gW + WOFF_Wv + (nt)     * 8 * TILE_ELEMS,
                 gW + WOFF_Wv + (nt + 1) * 8 * TILE_ELEMS, lane);
    acc_to_packed(sA1 + 8 * TILE_ELEMS, a0, nt * 16, false, lane);  // v -> sA1[8..15]
    acc_to_packed(sA1 + 8 * TILE_ELEMS, a1, (nt + 1) * 16, false, lane);
  }

  // ---- Stage 4: attended = LN(v @ Wo + bo) ----
  for (int p = 0; p < 8; ++p) {
    int nt = 2 * p;
    v8f a0 = splat8(bo[nt * 16 + nl]);
    v8f a1 = splat8(bo[(nt + 1) * 16 + nl]);
    gemm_acc2<8>(a0, a1, sA1 + 8 * TILE_ELEMS,
                 gW + WOFF_Wo + (nt)     * 8 * TILE_ELEMS,
                 gW + WOFF_Wo + (nt + 1) * 8 * TILE_ELEMS, lane);
    acc_to_sY(sY, a0, nt * 16, lane);
    acc_to_sY(sY, a1, (nt + 1) * 16, lane);
  }
  // attended -> sFus[8..15] and sA1[0..7] (first half of GRU input x)
  wave_ln_pack(sY, gn, bn, sFus + 8 * TILE_ELEMS, sA1, false, lane);

  // ---- Stage 5: stage prev_embed (x cols 256..511) and h (bf16) ----
  for (int idx = lane; idx < 4096; idx += 32) {
    int m = idx >> 8, c = idx & 255;
    int t = tok[R + m];
    (sA1 + 8 * TILE_ELEMS)[a_off(m, c)] = (__bf16)emb[(long)t * 256 + c];
  }
  pack_rows256(sA1 + 16 * TILE_ELEMS, ehs + (long)R * 256, lane);

  // ---- Stage 6: GRU, per 16-col tile: 4 accumulators, fused gating ----
  for (int t = 0; t < 16; ++t) {
    int c0 = t * 16;
    v8f ar  = splat8(bih[c0 + nl] + bhh[c0 + nl]);
    v8f az  = splat8(bih[256 + c0 + nl] + bhh[256 + c0 + nl]);
    v8f agn = splat8(bih[512 + c0 + nl]);
    v8f ahn = splat8(bhh[512 + c0 + nl]);
    const __bf16* pIr = gW + WOFF_Wih + (t)      * 16 * TILE_ELEMS;
    const __bf16* pIz = gW + WOFF_Wih + (16 + t) * 16 * TILE_ELEMS;
    const __bf16* pIn = gW + WOFF_Wih + (32 + t) * 16 * TILE_ELEMS;
    const __bf16* pHr = gW + WOFF_Whh + (t)      * 8 * TILE_ELEMS;
    const __bf16* pHz = gW + WOFF_Whh + (16 + t) * 8 * TILE_ELEMS;
    const __bf16* pHn = gW + WOFF_Whh + (32 + t) * 8 * TILE_ELEMS;
    for (int kt = 0; kt < 16; ++kt) {          // x = [attended|prev_embed], K=512
      AFrag a  = load_frag128(sA1 + kt * TILE_ELEMS, lane);
      AFrag b0 = load_frag128(pIr + kt * TILE_ELEMS, lane);
      AFrag b1 = load_frag128(pIz + kt * TILE_ELEMS, lane);
      AFrag b2 = load_frag128(pIn + kt * TILE_ELEMS, lane);
      ar  = __builtin_amdgcn_wmma_f32_16x16x32_bf16(false, a.v, false, b0.v, (short)0, ar,  false, false);
      az  = __builtin_amdgcn_wmma_f32_16x16x32_bf16(false, a.v, false, b1.v, (short)0, az,  false, false);
      agn = __builtin_amdgcn_wmma_f32_16x16x32_bf16(false, a.v, false, b2.v, (short)0, agn, false, false);
    }
    for (int kt = 0; kt < 8; ++kt) {           // h, K=256
      AFrag a  = load_frag128(sA1 + (16 + kt) * TILE_ELEMS, lane);
      AFrag b0 = load_frag128(pHr + kt * TILE_ELEMS, lane);
      AFrag b1 = load_frag128(pHz + kt * TILE_ELEMS, lane);
      AFrag b2 = load_frag128(pHn + kt * TILE_ELEMS, lane);
      ar  = __builtin_amdgcn_wmma_f32_16x16x32_bf16(false, a.v, false, b0.v, (short)0, ar,  false, false);
      az  = __builtin_amdgcn_wmma_f32_16x16x32_bf16(false, a.v, false, b1.v, (short)0, az,  false, false);
      ahn = __builtin_amdgcn_wmma_f32_16x16x32_bf16(false, a.v, false, b2.v, (short)0, ahn, false, false);
    }
#pragma unroll
    for (int r = 0; r < 8; ++r) {
      int m = r + half * 8;
      long grow = (long)(R + m);
      float hv = ehs[grow * 256 + c0 + nl];     // exact f32 h for the blend
      float rv = sigm(ar[r]);
      float zv = sigm(az[r]);
      float nv = tanhf(agn[r] + rv * ahn[r]);
      float nh = (1.f - zv) * nv + zv * hv;
      out_newh[grow * 256 + c0 + nl] = nh;
      sFus[a_off(m, 512 + c0 + nl)] = (__bf16)nh;   // new_h -> sFus[16..23]
    }
  }

  // ---- Stage 7: fused = relu(LN([edge|attended|new_h] @ Wf + bf)) ----
  for (int p = 0; p < 8; ++p) {
    int nt = 2 * p;
    v8f a0 = splat8(bfp[nt * 16 + nl]);
    v8f a1 = splat8(bfp[(nt + 1) * 16 + nl]);
    gemm_acc2<24>(a0, a1, sFus,
                  gW + WOFF_Wf + (nt)     * 24 * TILE_ELEMS,
                  gW + WOFF_Wf + (nt + 1) * 24 * TILE_ELEMS, lane);
    acc_to_sY(sY, a0, nt * 16, lane);
    acc_to_sY(sY, a1, (nt + 1) * 16, lane);
  }
  wave_ln_pack(sY, gf, bff, sA1, nullptr, true, lane);    // fused -> sA1[0..7]

  // ---- Stage 8: hidden = relu(fused @ W1 + b1), N=128 ----
  for (int p = 0; p < 4; ++p) {
    int nt = 2 * p;
    v8f a0 = splat8(b1[nt * 16 + nl]);
    v8f a1 = splat8(b1[(nt + 1) * 16 + nl]);
    gemm_acc2<8>(a0, a1, sA1,
                 gW + WOFF_W1 + (nt)     * 8 * TILE_ELEMS,
                 gW + WOFF_W1 + (nt + 1) * 8 * TILE_ELEMS, lane);
    acc_to_packed(sA1 + 8 * TILE_ELEMS, a0, nt * 16, true, lane); // -> sA1[8..11]
    acc_to_packed(sA1 + 8 * TILE_ELEMS, a1, (nt + 1) * 16, true, lane);
  }

  // ---- Stage 9: logits = hidden @ W2 + b2 (N padded to 16 in pack) ----
  {
    v8f acc = splat8((nl < 8) ? b2[nl] : 0.f);
    acc = gemm_acc<4>(acc, sA1 + 8 * TILE_ELEMS, gW + WOFF_W2, lane);
    if (nl < 8) {
#pragma unroll
      for (int r = 0; r < 8; ++r) {
        int m = r + half * 8;
        out_logits[(long)(R + m) * 8 + nl] = acc[r];
      }
    }
  }
}

// ------------------------------ host launch --------------------------------
static inline void pack_w(const float* src, int K, int N, int Kp, int Np,
                          int dstOff, hipStream_t stream) {
  int total = Kp * Np;
  pack_weight_kernel<<<(total + 255) / 256, 256, 0, stream>>>(src, K, N, Kp, Np, dstOff);
}

extern "C" void kernel_launch(void* const* d_in, const int* in_sizes, int n_in,
                              void* d_out, int out_size, void* d_ws, size_t ws_size,
                              hipStream_t stream) {
  const float* node_i = (const float*)d_in[0];
  const float* node_j = (const float*)d_in[1];
  const float* latent = (const float*)d_in[2];
  const float* ehs    = (const float*)d_in[3];
  const int*   tok    = (const int*)d_in[4];
  const float* We  = (const float*)d_in[5];
  const float* be  = (const float*)d_in[6];
  const float* ge  = (const float*)d_in[7];
  const float* bee = (const float*)d_in[8];
  const float* Wc  = (const float*)d_in[9];
  const float* bc  = (const float*)d_in[10];
  // d_in[11..14] = Wq,bq,Wk,bk : mathematically inert (single-key attention)
  const float* Wv  = (const float*)d_in[15];
  const float* bvv = (const float*)d_in[16];
  const float* Wo  = (const float*)d_in[17];
  const float* bo  = (const float*)d_in[18];
  const float* gn  = (const float*)d_in[19];
  const float* bn  = (const float*)d_in[20];
  const float* emb = (const float*)d_in[21];
  const float* Wih = (const float*)d_in[22];
  const float* bih = (const float*)d_in[23];
  const float* Whh = (const float*)d_in[24];
  const float* bhh = (const float*)d_in[25];
  const float* Wf  = (const float*)d_in[26];
  const float* bfp = (const float*)d_in[27];
  const float* gf  = (const float*)d_in[28];
  const float* bff = (const float*)d_in[29];
  const float* W1  = (const float*)d_in[30];
  const float* b1  = (const float*)d_in[31];
  const float* W2  = (const float*)d_in[32];
  const float* b2  = (const float*)d_in[33];

  // Pack every GEMM B operand into bf16 fragment layout (deterministic, cheap).
  pack_w(We,  512, 256, 512, 256, WOFF_We,  stream);
  pack_w(Wc,    8, 256,  32, 256, WOFF_Wc,  stream);   // K zero-padded
  pack_w(Wv,  256, 256, 256, 256, WOFF_Wv,  stream);
  pack_w(Wo,  256, 256, 256, 256, WOFF_Wo,  stream);
  pack_w(Wih, 512, 768, 512, 768, WOFF_Wih, stream);
  pack_w(Whh, 256, 768, 256, 768, WOFF_Whh, stream);
  pack_w(Wf,  768, 256, 768, 256, WOFF_Wf,  stream);
  pack_w(W1,  256, 128, 256, 128, WOFF_W1,  stream);
  pack_w(W2,  128,   8, 128,  16, WOFF_W2,  stream);   // N zero-padded

  int Bn = in_sizes[0] / 256;                 // B = 65536
  float* out = (float*)d_out;
  float* out_logits = out;                    // [B, 8]
  float* out_newh   = out + (size_t)Bn * 8;   // [B, 256]

  dim3 grid((unsigned)((Bn + 63) / 64)), block(128);
  size_t shmem = 4 * (size_t)WAVE_SLICE;      // 260 KB (<= 320 KB/WGP)
  edge_decoder_kernel<<<grid, block, shmem, stream>>>(
      node_i, node_j, latent, ehs, tok,
      be, ge, bee, bc, bvv, bo, gn, bn, emb,
      bih, bhh, bfp, gf, bff, b1, b2,
      out_logits, out_newh, Bn);
}
